// CycleS_24154896073303
// MI455X (gfx1250) — matrix operand order repaired and checked by reading, not compiled, not run on hardware
//
#include <hip/hip_runtime.h>
#include <hip/hip_bf16.h>

// ---------------------------------------------------------------------------
// CDNA5 (gfx1250) implementation of the cycle-consistency loss.
// All big GEMMs run on v_wmma_f32_16x16x32_bf16 (bf16 in, fp32 accumulate),
// with a double-buffered LDS software pipeline to overlap HBM/L2 fetch with
// matrix-core work, plus global_prefetch_b8 one stage ahead.
// ---------------------------------------------------------------------------

typedef __attribute__((ext_vector_type(16))) __bf16 v16bf;
typedef __attribute__((ext_vector_type(8)))  __bf16 v8bf;
typedef __attribute__((ext_vector_type(8)))  float  v8f;

static constexpr int   NDIM    = 4096;
static constexpr int   DDIM    = 1024;
static constexpr int   CPART   = 16;                     // column-softmax row chunks
static constexpr float SMSCALE = 83.17766166719343f;     // log(4096)/0.1
static constexpr float MARGIN  = 0.5f;

__device__ __forceinline__ __bf16 f2bf(float f) {
    union { float f; unsigned u; } in; in.f = f;
    unsigned u = in.u;
    unsigned r = (u + 0x7FFFu + ((u >> 16) & 1u)) >> 16;   // round-to-nearest-even
    union { unsigned short s; __bf16 h; } out;
    out.s = (unsigned short)r;
    return out.h;
}

// ---------------------------------------------------------------------------
// Row L2-normalize fp32 -> bf16
// ---------------------------------------------------------------------------
__global__ __launch_bounds__(256)
void normalize_rows_k(const float* __restrict__ X, __bf16* __restrict__ Y) {
    __shared__ float red[256];
    const int row = blockIdx.x;
    const float* x = X + (size_t)row * DDIM;
    float ss = 0.0f;
    for (int i = threadIdx.x; i < DDIM; i += 256) { float v = x[i]; ss += v * v; }
    red[threadIdx.x] = ss; __syncthreads();
    for (int s = 128; s > 0; s >>= 1) {
        if (threadIdx.x < s) red[threadIdx.x] += red[threadIdx.x + s];
        __syncthreads();
    }
    const float scale = 1.0f / fmaxf(sqrtf(red[0]), 1e-12f);
    __bf16* y = Y + (size_t)row * DDIM;
    for (int i = threadIdx.x; i < DDIM; i += 256) y[i] = f2bf(x[i] * scale);
}

// ---------------------------------------------------------------------------
// Tiled WMMA GEMM:  C[M,N] = A[M,K](bf16) @ op(B)(bf16), fp32 accumulate.
//   BT=true : B stored [N,K] (NT GEMM) ; BT=false : B stored [K,N] (NN GEMM)
//   OUT_F32 / OUT_BF16 : write C / Cb
//   FUSED   : don't store C; write diag and atomically reduce off-diag
//             row-max (rmaxv) / col-max (cmaxv).  (Entries >= 0.)
//
// Block = 256 threads = 8 waves; tile 128x128, BK=32; wave tile 64x32
// (4x2 of 16x16x32 WMMA).  LDS row pitch 48 bf16 (96B) keeps every fragment
// chunk 16B-aligned -> ds_load_b128.
//
// Software pipeline: registers stage tile kt+32 while WMMAs consume the LDS
// ping-pong buffer holding tile kt; tile kt+64 is prefetched into L2.
// ---------------------------------------------------------------------------
template<bool BT, bool OUT_F32, bool OUT_BF16, bool FUSED>
__global__ __launch_bounds__(256)
void gemm_wmma_k(const __bf16* __restrict__ A, int lda,
                 const __bf16* __restrict__ B, int ldb,
                 float* __restrict__ C, __bf16* __restrict__ Cb, int ldc,
                 int K,
                 float* __restrict__ dvec, float* __restrict__ rmaxv,
                 float* __restrict__ cmaxv)
{
    __shared__ __align__(16) __bf16 lA[2][128 * 48];
    __shared__ __align__(16) __bf16 lB[2][128 * 48];

    const int tid   = threadIdx.x;
    const int wave  = tid >> 5;        // wave32
    const int lane  = tid & 31;
    const int lhalf = lane >> 4;
    const int lmod  = lane & 15;
    const int wm    = (wave >> 2) * 64;
    const int wn    = (wave & 3) * 32;
    const int mBase = blockIdx.y * 128;
    const int nBase = blockIdx.x * 128;

    // staging coordinates (two 16B chunks per thread per matrix)
    const int c0 = tid, c1 = tid + 256;
    const int ar0 = c0 >> 2, ac0 = (c0 & 3) << 3;   // A (and BT-B): [row][col8]
    const int ar1 = c1 >> 2, ac1 = (c1 & 3) << 3;
    const int bk0 = c0 & 31, bn0 = (c0 >> 5) << 3;  // NN-B: [k][n8]
    const int bk1 = c1 & 31, bn1 = (c1 >> 5) << 3;

    uint4 ra0, ra1, rb0, rb1;
    auto stage = [&](int kt) {
        ra0 = *(const uint4*)&A[(size_t)(mBase + ar0) * lda + kt + ac0];
        ra1 = *(const uint4*)&A[(size_t)(mBase + ar1) * lda + kt + ac1];
        if (BT) {
            rb0 = *(const uint4*)&B[(size_t)(nBase + ar0) * ldb + kt + ac0];
            rb1 = *(const uint4*)&B[(size_t)(nBase + ar1) * ldb + kt + ac1];
        } else {
            rb0 = *(const uint4*)&B[(size_t)(kt + bk0) * ldb + nBase + bn0];
            rb1 = *(const uint4*)&B[(size_t)(kt + bk1) * ldb + nBase + bn1];
        }
    };
    auto commit = [&](int buf) {
        *(uint4*)&lA[buf][ar0 * 48 + ac0] = ra0;
        *(uint4*)&lA[buf][ar1 * 48 + ac1] = ra1;
        if (BT) {
            *(uint4*)&lB[buf][ar0 * 48 + ac0] = rb0;
            *(uint4*)&lB[buf][ar1 * 48 + ac1] = rb1;
        } else {
            union { uint4 u; __bf16 h[8]; } w;
            w.u = rb0;
            #pragma unroll
            for (int e = 0; e < 8; ++e) lB[buf][(bn0 + e) * 48 + bk0] = w.h[e];
            w.u = rb1;
            #pragma unroll
            for (int e = 0; e < 8; ++e) lB[buf][(bn1 + e) * 48 + bk1] = w.h[e];
        }
    };

    v8f acc[4][2];
    #pragma unroll
    for (int i = 0; i < 4; ++i)
        #pragma unroll
        for (int j = 0; j < 2; ++j)
            #pragma unroll
            for (int e = 0; e < 8; ++e) acc[i][j][e] = 0.0f;

    stage(0);
    int buf = 0;
    for (int kt = 0; kt < K; kt += 32) {
        commit(buf);
        __syncthreads();

        if (kt + 32 < K) {
            stage(kt + 32);                    // global loads overlap WMMAs below
            if (kt + 64 < K) {                 // pre-warm L2 one stage further
                __builtin_prefetch(&A[(size_t)(mBase + ar0) * lda + kt + 64 + ac0], 0, 0);
                if (BT)
                    __builtin_prefetch(&B[(size_t)(nBase + ar0) * ldb + kt + 64 + ac0], 0, 0);
                else
                    __builtin_prefetch(&B[(size_t)(kt + 64 + bk0) * ldb + nBase + bn0], 0, 0);
            }
        }

        // A fragments: lane m = sub + lmod; K elems {lhalf*8+0..7, 16+lhalf*8+0..7}
        v16bf afrag[4];
        #pragma unroll
        for (int s = 0; s < 4; ++s) {
            int m = wm + s * 16 + lmod;
            union { uint4 u[2]; v16bf v; } fr;
            fr.u[0] = *(const uint4*)&lA[buf][m * 48 + lhalf * 8];
            fr.u[1] = *(const uint4*)&lA[buf][m * 48 + 16 + lhalf * 8];
            afrag[s] = fr.v;
        }
        // B fragments: lane n = sub + lmod; K elems {lhalf*8+0..7, 16+lhalf*8+0..7}
        v16bf bfrag[2];
        #pragma unroll
        for (int s = 0; s < 2; ++s) {
            int n = wn + s * 16 + lmod;
            union { uint4 u[2]; v16bf v; } fr;
            fr.u[0] = *(const uint4*)&lB[buf][n * 48 + lhalf * 8];
            fr.u[1] = *(const uint4*)&lB[buf][n * 48 + 16 + lhalf * 8];
            bfrag[s] = fr.v;
        }
        #pragma unroll
        for (int sm = 0; sm < 4; ++sm)
            #pragma unroll
            for (int sn = 0; sn < 2; ++sn)
                acc[sm][sn] = __builtin_amdgcn_wmma_f32_16x16x32_bf16(
                    false, afrag[sm], false, bfrag[sn],
                    (short)0, acc[sm][sn], false, false);

        buf ^= 1;
        __syncthreads();   // protect lds[buf] (rewritten two iterations later)
    }

    if (FUSED) {
        // diag + per-row max (shuffle-reduce across the 16 lanes sharing a row)
        #pragma unroll
        for (int sm = 0; sm < 4; ++sm) {
            #pragma unroll
            for (int r = 0; r < 8; ++r) {
                int m = mBase + wm + sm * 16 + lhalf * 8 + r;
                float rc = 0.0f;
                #pragma unroll
                for (int sn = 0; sn < 2; ++sn) {
                    int n = nBase + wn + sn * 16 + lmod;
                    float v = acc[sm][sn][r];
                    if (m == n) { dvec[m] = v; v = 0.0f; }  // exclude diagonal
                    rc = fmaxf(rc, v);
                }
                #pragma unroll
                for (int o = 8; o >= 1; o >>= 1)
                    rc = fmaxf(rc, __shfl_xor(rc, o, 32));
                if (lmod == 0)
                    atomicMax((unsigned int*)&rmaxv[m], __float_as_uint(rc));
            }
        }
        // per-column max (lane-local reduce then one atomic per column slice)
        #pragma unroll
        for (int sn = 0; sn < 2; ++sn) {
            int n = nBase + wn + sn * 16 + lmod;
            float cc = 0.0f;
            #pragma unroll
            for (int sm = 0; sm < 4; ++sm)
                #pragma unroll
                for (int r = 0; r < 8; ++r) {
                    int m = mBase + wm + sm * 16 + lhalf * 8 + r;
                    float v = acc[sm][sn][r];
                    cc = fmaxf(cc, (m == n) ? 0.0f : v);
                }
            atomicMax((unsigned int*)&cmaxv[n], __float_as_uint(cc));
        }
    } else {
        #pragma unroll
        for (int sm = 0; sm < 4; ++sm)
            #pragma unroll
            for (int sn = 0; sn < 2; ++sn)
                #pragma unroll
                for (int r = 0; r < 8; ++r) {
                    int m = mBase + wm + sm * 16 + lhalf * 8 + r;
                    int n = nBase + wn + sn * 16 + lmod;
                    float v = acc[sm][sn][r];
                    if (OUT_F32)  C [(size_t)m * ldc + n] = v;
                    if (OUT_BF16) Cb[(size_t)m * ldc + n] = f2bf(v);
                }
    }
}

// ---------------------------------------------------------------------------
// Softmax pipeline
// ---------------------------------------------------------------------------
__global__ __launch_bounds__(256)
void row_softmax_k(const float* __restrict__ S, __bf16* __restrict__ O) {
    __shared__ float red[256];
    const int row = blockIdx.x;
    const float* s = S + (size_t)row * NDIM;
    float vals[NDIM / 256];
    float mx = -__builtin_inff();
    #pragma unroll
    for (int e = 0; e < NDIM / 256; ++e) {
        vals[e] = s[e * 256 + threadIdx.x] * SMSCALE;
        mx = fmaxf(mx, vals[e]);
    }
    red[threadIdx.x] = mx; __syncthreads();
    for (int st = 128; st > 0; st >>= 1) {
        if (threadIdx.x < st) red[threadIdx.x] = fmaxf(red[threadIdx.x], red[threadIdx.x + st]);
        __syncthreads();
    }
    mx = red[0]; __syncthreads();
    float sum = 0.0f;
    #pragma unroll
    for (int e = 0; e < NDIM / 256; ++e) { vals[e] = __expf(vals[e] - mx); sum += vals[e]; }
    red[threadIdx.x] = sum; __syncthreads();
    for (int st = 128; st > 0; st >>= 1) {
        if (threadIdx.x < st) red[threadIdx.x] += red[threadIdx.x + st];
        __syncthreads();
    }
    const float inv = 1.0f / red[0];
    __bf16* o = O + (size_t)row * NDIM;
    #pragma unroll
    for (int e = 0; e < NDIM / 256; ++e)
        o[e * 256 + threadIdx.x] = f2bf(vals[e] * inv);
}

// column softmax: online max/sum over a chunk of rows (coalesced)
__global__ __launch_bounds__(256)
void col_partial_k(const float* __restrict__ S, float* __restrict__ pm,
                   float* __restrict__ ps) {
    const int c = blockIdx.x * 256 + threadIdx.x;
    const int chunk = blockIdx.y;
    const int r0 = chunk * (NDIM / CPART);
    float m = -__builtin_inff(), s = 0.0f;
    for (int r = r0; r < r0 + NDIM / CPART; ++r) {
        float v = S[(size_t)r * NDIM + c] * SMSCALE;
        if (v > m) { s = s * __expf(m - v) + 1.0f; m = v; }
        else       { s += __expf(v - m); }
    }
    pm[(size_t)chunk * NDIM + c] = m;
    ps[(size_t)chunk * NDIM + c] = s;
}

__global__ __launch_bounds__(256)
void col_combine_k(const float* __restrict__ pm, const float* __restrict__ ps,
                   float* __restrict__ cm, float* __restrict__ cs) {
    const int c = blockIdx.x * 256 + threadIdx.x;
    float m = -__builtin_inff();
    #pragma unroll
    for (int p = 0; p < CPART; ++p) m = fmaxf(m, pm[(size_t)p * NDIM + c]);
    float s = 0.0f;
    #pragma unroll
    for (int p = 0; p < CPART; ++p)
        s += ps[(size_t)p * NDIM + c] * __expf(pm[(size_t)p * NDIM + c] - m);
    cm[c] = m; cs[c] = s;
}

// write column-softmax result (same [r][c] layout; consumed as B^T in NT GEMM)
__global__ __launch_bounds__(256)
void col_normalize_k(const float* __restrict__ S, const float* __restrict__ cm,
                     const float* __restrict__ cs, __bf16* __restrict__ O) {
    const size_t idx = (size_t)blockIdx.x * 256 + threadIdx.x;
    const int c = (int)(idx & (NDIM - 1));
    O[idx] = f2bf(__expf(S[idx] * SMSCALE - cm[c]) / cs[c]);
}

// ---------------------------------------------------------------------------
// Misc
// ---------------------------------------------------------------------------
__global__ void zero_f32_k(float* p, int n) {
    int i = blockIdx.x * blockDim.x + threadIdx.x;
    if (i < n) p[i] = 0.0f;
}

__global__ __launch_bounds__(256)
void hinge_accum_k(const float* __restrict__ dvec, const float* __restrict__ rmaxv,
                   const float* __restrict__ cmaxv, float* __restrict__ out,
                   float weight) {
    __shared__ float red[256];
    float acc = 0.0f;
    for (int i = threadIdx.x; i < NDIM; i += 256) {
        float d = dvec[i];
        acc += fmaxf(rmaxv[i] + MARGIN - d, 0.0f);
        acc += fmaxf(cmaxv[i] + MARGIN - d, 0.0f);
    }
    red[threadIdx.x] = acc; __syncthreads();
    for (int s = 128; s > 0; s >>= 1) {
        if (threadIdx.x < s) red[threadIdx.x] += red[threadIdx.x + s];
        __syncthreads();
    }
    if (threadIdx.x == 0)
        out[0] += weight * red[0] / (2.0f * (float)NDIM);
}

// ---------------------------------------------------------------------------
// Orchestration
// ---------------------------------------------------------------------------
extern "C" void kernel_launch(void* const* d_in, const int* in_sizes, int n_in,
                              void* d_out, int out_size, void* d_ws, size_t ws_size,
                              hipStream_t stream) {
    const float* feat[3] = { (const float*)d_in[0], (const float*)d_in[1],
                             (const float*)d_in[2] };
    float* out = (float*)d_out;

    // workspace carve (256B aligned): ~160 MB total
    char* ws = (char*)d_ws;
    size_t off = 0;
    auto carve = [&](size_t bytes) -> void* {
        void* p = ws + off;
        off += (bytes + 255) & ~(size_t)255;
        return p;
    };
    __bf16* nfb[3];
    for (int v = 0; v < 3; ++v) nfb[v] = (__bf16*)carve((size_t)NDIM * DDIM * 2);
    float*  bufS  = (float*) carve((size_t)NDIM * NDIM * 4);   // fp32 similarity
    __bf16* bufA  = (__bf16*)carve((size_t)NDIM * NDIM * 2);   // S_ik / Ahat
    __bf16* bufB  = (__bf16*)carve((size_t)NDIM * NDIM * 2);   // S_kj / Bhat
    float*  colpm = (float*) carve((size_t)CPART * NDIM * 4);
    float*  colps = (float*) carve((size_t)CPART * NDIM * 4);
    float*  colm  = (float*) carve((size_t)NDIM * 4);
    float*  cols  = (float*) carve((size_t)NDIM * 4);
    float*  dvec  = (float*) carve((size_t)NDIM * 4);
    float*  rmaxv = (float*) carve((size_t)NDIM * 4);
    float*  cmaxv = (float*) carve((size_t)NDIM * 4);

    const dim3 gemmGrid(NDIM / 128, NDIM / 128);   // 32x32 blocks
    const dim3 blk(256);

    zero_f32_k<<<1, 32, 0, stream>>>(out, out_size);

    for (int v = 0; v < 3; ++v)
        normalize_rows_k<<<NDIM, blk, 0, stream>>>(feat[v], nfb[v]);

    // cycle term on the fp32 S currently in bufS; adds weight/3 into out[0]
    auto cycle_term = [&]() {
        row_softmax_k<<<NDIM, blk, 0, stream>>>(bufS, bufA);
        col_partial_k<<<dim3(NDIM / 256, CPART), blk, 0, stream>>>(bufS, colpm, colps);
        col_combine_k<<<NDIM / 256, blk, 0, stream>>>(colpm, colps, colm, cols);
        col_normalize_k<<<(unsigned)((size_t)NDIM * NDIM / 256), blk, 0, stream>>>(
            bufS, colm, cols, bufB);
        zero_f32_k<<<NDIM / 256, blk, 0, stream>>>(rmaxv, NDIM);
        zero_f32_k<<<NDIM / 256, blk, 0, stream>>>(cmaxv, NDIM);
        // M = Ahat @ Bhat^T, fused diag/rowmax/colmax reduction (no M store)
        gemm_wmma_k<true, false, false, true><<<gemmGrid, blk, 0, stream>>>(
            bufA, NDIM, bufB, NDIM, nullptr, nullptr, NDIM, NDIM,
            dvec, rmaxv, cmaxv);
        hinge_accum_k<<<1, blk, 0, stream>>>(dvec, rmaxv, cmaxv, out, 1.0f / 3.0f);
    };

    // ---- pairwise terms: S = nf_i @ nf_j^T (NT, K=1024, fp32 out) ----
    const int pairs[3][2] = { {0, 1}, {0, 2}, {1, 2} };
    for (int p = 0; p < 3; ++p) {
        gemm_wmma_k<true, true, false, false><<<gemmGrid, blk, 0, stream>>>(
            nfb[pairs[p][0]], DDIM, nfb[pairs[p][1]], DDIM,
            bufS, nullptr, NDIM, DDIM, nullptr, nullptr, nullptr);
        cycle_term();
    }

    // ---- triple terms: S = (nf_i nf_k^T)(nf_k nf_j^T), K=4096 ----
    const int tri[3][3] = { {0, 1, 2}, {0, 2, 1}, {1, 2, 0} };  // (i,j,k)
    for (int t = 0; t < 3; ++t) {
        const int i = tri[t][0], j = tri[t][1], k = tri[t][2];
        gemm_wmma_k<true, false, true, false><<<gemmGrid, blk, 0, stream>>>(
            nfb[i], DDIM, nfb[k], DDIM, nullptr, bufA, NDIM, DDIM,
            nullptr, nullptr, nullptr);                          // S_ik (bf16)
        gemm_wmma_k<true, false, true, false><<<gemmGrid, blk, 0, stream>>>(
            nfb[k], DDIM, nfb[j], DDIM, nullptr, bufB, NDIM, DDIM,
            nullptr, nullptr, nullptr);                          // S_kj (bf16)
        gemm_wmma_k<false, true, false, false><<<gemmGrid, blk, 0, stream>>>(
            bufA, NDIM, bufB, NDIM, bufS, nullptr, NDIM, NDIM,
            nullptr, nullptr, nullptr);                          // S_tri (fp32)
        cycle_term();
    }

    (void)in_sizes; (void)n_in; (void)ws_size;
}
